// EmbedMatcher_75840532512960
// MI455X (gfx1250) — matrix operand order repaired and checked by reading, not compiled, not run on hardware
//
#include <hip/hip_runtime.h>
#include <hip/hip_bf16.h>

typedef _Float16 v8h  __attribute__((ext_vector_type(8)));
typedef _Float16 v16h __attribute__((ext_vector_type(16)));
typedef float    v8f  __attribute__((ext_vector_type(8)));

#define B_ROWS 4096
#define FEW    5
#define NNB    64
#define DE     128   // embed dim
#define DM     256   // d_model
#define HDIM   512   // LSTM hidden
#define KTOT   768   // [q(256) | h(256) | r(256)]
#define NGATE  2048  // 4*HDIM

__device__ __forceinline__ float sigmf(float x) { return 1.0f / (1.0f + __expf(-x)); }

static __device__ __forceinline__ v16h mkfrag(v8h lo, v8h hi) {
    return __builtin_shufflevector(lo, hi, 0,1,2,3,4,5,6,7,8,9,10,11,12,13,14,15);
}

// ---------------------------------------------------------------------------
// Prep: hop-gate softmax, v = gcn_w^T @ attn_w^T, fused biases
// ---------------------------------------------------------------------------
__global__ void __launch_bounds__(256) prep_kernel(
    const float* __restrict__ hop_gate, const float* __restrict__ attn_w,
    const float* __restrict__ gcn_w, const float* __restrict__ gcn_wb,
    const float* __restrict__ gcn_b, const float* __restrict__ b_ih,
    const float* __restrict__ b_hh,
    float* __restrict__ gwv, float* __restrict__ vvec,
    float* __restrict__ gbias, float* __restrict__ biasg)
{
    const int t = threadIdx.x;
    if (t == 0) {
        float a = hop_gate[0], b = hop_gate[1];
        float m = fmaxf(a, b);
        float e0 = __expf(a - m), e1 = __expf(b - m);
        gwv[0] = e0 / (e0 + e1);
        gwv[1] = e1 / (e0 + e1);
    }
    float acc = 0.f;
    for (int d = 0; d < DE; ++d) acc += attn_w[d] * gcn_w[d * DM + t];
    vvec[t] = acc;
    if (t < DE) gbias[t] = gcn_wb[t] + gcn_b[t];
    for (int j = t; j < NGATE; j += 256) biasg[j] = b_ih[j] + b_hh[j];
}

// ---------------------------------------------------------------------------
// Convert [w_ih | w_hh] -> f16 Wc[2048][768]
// ---------------------------------------------------------------------------
__global__ void __launch_bounds__(256) convw_kernel(
    const float* __restrict__ w_ih, const float* __restrict__ w_hh,
    _Float16* __restrict__ Wc)
{
    int idx = blockIdx.x * 256 + threadIdx.x;
    if (idx >= NGATE * KTOT) return;
    int j = idx / KTOT, k = idx % KTOT;
    float v = (k < DM) ? w_ih[j * DM + k] : w_hh[j * HDIM + (k - DM)];
    Wc[idx] = (_Float16)v;
}

// ---------------------------------------------------------------------------
// Fused neighbor encoder: gather + score matvec + softmax + weighted sum
// + gcn matvec + tanh + hop-gate blend.  One block per (row, side).
// ---------------------------------------------------------------------------
__global__ void __launch_bounds__(256) neighbor_kernel(
    const int* __restrict__ q_l1, const int* __restrict__ q_l2,
    const int* __restrict__ q_r1, const int* __restrict__ q_r2,
    const int* __restrict__ s_l1, const int* __restrict__ s_l2,
    const int* __restrict__ s_r1, const int* __restrict__ s_r2,
    const float* __restrict__ emb, const float* __restrict__ gcn_w,
    const float* __restrict__ gbias, const float* __restrict__ vvec,
    const float* __restrict__ gwv,
    float* __restrict__ QN, float* __restrict__ SN,
    _Float16* __restrict__ Xa, _Float16* __restrict__ Xb)
{
    __shared__ _Float16 xs[NNB][DM];   // gathered concat values (f16, 32KB)
    __shared__ float vsh[DM];
    __shared__ float scores[NNB];
    __shared__ float wts[NNB];
    __shared__ float wsum[DM];

    const int t = threadIdx.x;
    vsh[t] = vvec[t];

    const int task = blockIdx.x;
    const int side = task & 1;
    const int idx  = task >> 1;
    const bool isQ = idx < B_ROWS;
    const int row  = isQ ? idx : idx - B_ROWS;

    const int* conns[8] = { q_l1, q_l2, q_r1, q_r2, s_l1, s_l2, s_r1, s_r2 };
    const float g0 = gwv[0], g1 = gwv[1];

    const int wid  = t >> 5;
    const int lane = t & 31;
    const bool isRel = lane < 16;
    const int lsub = isRel ? lane : lane - 16;

    float res = 0.f;   // threads 0..127 accumulate proj dim d = t

    for (int hop = 0; hop < 2; ++hop) {
        __syncthreads();
        const int* conn = conns[(isQ ? 0 : 4) + side * 2 + hop];
        // gather + per-neighbor score dot: wave `wid` owns 8 neighbors
        for (int k = 0; k < 8; ++k) {
            const int nn = wid * 8 + k;
            const int cidx = (row * NNB + nn) * 2;
            const int sym = conn[cidx + (isRel ? 0 : 1)];
            const float* ep = emb + (size_t)sym * DE + lsub * 8;
            const int cb = (isRel ? 0 : DE) + lsub * 8;
            float partial = 0.f;
#pragma unroll
            for (int i = 0; i < 8; ++i) {
                float val = ep[i];
                partial += val * vsh[cb + i];
                xs[nn][cb + i] = (_Float16)val;
            }
#pragma unroll
            for (int off = 16; off; off >>= 1) partial += __shfl_xor(partial, off);
            if (lane == 0) scores[nn] = partial;
        }
        __syncthreads();
        // softmax over 64 scores (one wave)
        if (t < 32) {
            float sa = scores[t], sb = scores[t + 32];
            float mx = fmaxf(sa, sb);
#pragma unroll
            for (int off = 16; off; off >>= 1) mx = fmaxf(mx, __shfl_xor(mx, off));
            float ea = __expf(sa - mx), eb = __expf(sb - mx);
            float sm = ea + eb;
#pragma unroll
            for (int off = 16; off; off >>= 1) sm += __shfl_xor(sm, off);
            wts[t] = ea / sm;
            wts[t + 32] = eb / sm;
        }
        __syncthreads();
        // weighted sum over neighbors, channel t
        float acc = 0.f;
#pragma unroll 8
        for (int nn = 0; nn < NNB; ++nn) acc += wts[nn] * (float)xs[nn][t];
        wsum[t] = acc;
        __syncthreads();
        // proj matvec + tanh + hop blend (threads 0..127)
        if (t < DE) {
            float p = gbias[t];
            const float* gr = gcn_w + t * DM;
            for (int c = 0; c < DM; ++c) p += wsum[c] * gr[c];
            res += (hop == 0 ? g0 : g1) * tanhf(p);
        }
    }
    __syncthreads();
    if (t < DE) {
        if (isQ) {
            QN[row * DM + side * DE + t] = res;
            Xa[row * KTOT + side * DE + t] = (_Float16)res;
            Xb[row * KTOT + side * DE + t] = (_Float16)res;
        } else {
            SN[row * DM + side * DE + t] = res;
        }
    }
    if (isQ && side == 0) {   // zero h/r tail of step-0 operand buffer
        Xa[row * KTOT + DM + 2 * t]     = (_Float16)0.f;
        Xa[row * KTOT + DM + 2 * t + 1] = (_Float16)0.f;
    }
}

// ---------------------------------------------------------------------------
// Support encoder: MLP + residual + LayerNorm, FEW=5 rows (1 block)
// ---------------------------------------------------------------------------
__global__ void __launch_bounds__(256) support_enc_kernel(
    const float* __restrict__ SN,
    const float* __restrict__ p1_w, const float* __restrict__ p1_b,
    const float* __restrict__ p2_w, const float* __restrict__ p2_b,
    const float* __restrict__ ln_g, const float* __restrict__ ln_b,
    float* __restrict__ SG, float* __restrict__ SM)
{
    __shared__ float snr[DM];
    __shared__ float hsh[2 * DM];
    __shared__ float red1[256];
    __shared__ float red2[256];
    const int t = threadIdx.x;
    float msum = 0.f;
    for (int s = 0; s < FEW; ++s) {
        __syncthreads();
        snr[t] = SN[s * DM + t];
        __syncthreads();
        for (int jj = 0; jj < 2; ++jj) {
            int j = t + jj * 256;
            float a = p1_b[j];
            for (int c = 0; c < DM; ++c) a += snr[c] * p1_w[j * DM + c];
            hsh[j] = a > 0.f ? a : 0.f;
        }
        __syncthreads();
        float z = p2_b[t];
        for (int j = 0; j < 2 * DM; ++j) z += hsh[j] * p2_w[t * (2 * DM) + j];
        z += snr[t];
        red1[t] = z;
        red2[t] = z * z;
        __syncthreads();
        for (int stp = 128; stp > 0; stp >>= 1) {
            if (t < stp) { red1[t] += red1[t + stp]; red2[t] += red2[t + stp]; }
            __syncthreads();
        }
        float mu  = red1[0] / 256.f;
        float var = red2[0] / 256.f - mu * mu;
        float g = (z - mu) * rsqrtf(var + 1e-6f) * ln_g[t] + ln_b[t];
        SG[s * DM + t] = g;
        msum += g;
    }
    SM[t] = msum / (float)FEW;
}

// ---------------------------------------------------------------------------
// WMMA GEMM (gates = X @ Wc^T + bias) fused with LSTM cell + h update.
// Unrolled-by-2 ping-pong double buffer: each WMMA group consumes the
// fragment registers its loads wrote (no rotation movs), while the other
// buffer's loads are in flight.
// ---------------------------------------------------------------------------
__device__ __forceinline__ v16h ldfrag(const _Float16* p, int k0, int o0, int o1) {
    return mkfrag(*(const v8h*)(p + k0 + o0), *(const v8h*)(p + k0 + o1));
}

__global__ void __launch_bounds__(256) gemm_lstm_kernel(
    const _Float16* __restrict__ X, const _Float16* __restrict__ W,
    const float* __restrict__ biasg, float* __restrict__ Cst,
    const float* __restrict__ QN, float* __restrict__ Hout,
    _Float16* __restrict__ Xn, int step)
{
    const int wid   = threadIdx.x >> 5;
    const int lane  = threadIdx.x & 31;
    const int task  = blockIdx.x * 8 + wid;      // 8192 wave tasks
    const int mTile = task >> 5;                 // 256 m-tiles
    const int jt    = task & 31;                 // 32 j-tiles (512/16)
    const int mn    = lane & 15;
    const int khalf = lane >> 4;

    const _Float16* arow = X + (mTile * 16 + mn) * KTOT;
    const _Float16* b0 = W + (0 * HDIM + jt * 16 + mn) * KTOT;
    const _Float16* b1 = W + (1 * HDIM + jt * 16 + mn) * KTOT;
    const _Float16* b2 = W + (2 * HDIM + jt * 16 + mn) * KTOT;
    const _Float16* b3 = W + (3 * HDIM + jt * 16 + mn) * KTOT;

    v8f acc0 = {0.f,0.f,0.f,0.f,0.f,0.f,0.f,0.f};
    v8f acc1 = acc0, acc2 = acc0, acc3 = acc0;

    const int aoff0 = khalf * 8;        // A frag: interleaved K layout
    const int aoff1 = 16 + khalf * 8;
    const int boff0 = khalf * 8;        // B frag: interleaved K layout
    const int boff1 = 16 + khalf * 8;

    constexpr int NK = KTOT / 32;       // 24 k-tiles (even)

#define WMMA4(A, F0, F1, F2, F3)                                                                        \
    acc0 = __builtin_amdgcn_wmma_f32_16x16x32_f16(false, A, false, F0, (short)0, acc0, false, false);   \
    acc1 = __builtin_amdgcn_wmma_f32_16x16x32_f16(false, A, false, F1, (short)0, acc1, false, false);   \
    acc2 = __builtin_amdgcn_wmma_f32_16x16x32_f16(false, A, false, F2, (short)0, acc2, false, false);   \
    acc3 = __builtin_amdgcn_wmma_f32_16x16x32_f16(false, A, false, F3, (short)0, acc3, false, false);

    // stage tile 0 -> buffer 0
    v16h a0 = ldfrag(arow, 0, aoff0, aoff1);
    v16h p0 = ldfrag(b0, 0, boff0, boff1);
    v16h p1 = ldfrag(b1, 0, boff0, boff1);
    v16h p2 = ldfrag(b2, 0, boff0, boff1);
    v16h p3 = ldfrag(b3, 0, boff0, boff1);
    v16h a1, q0, q1, q2, q3;

#pragma unroll 1
    for (int kt = 0; kt < NK - 2; kt += 2) {
        const int ka = (kt + 1) * 32;
        const int kb = (kt + 2) * 32;
        // prefetch tile kt+1 -> buffer 1, compute on buffer 0
        a1 = ldfrag(arow, ka, aoff0, aoff1);
        q0 = ldfrag(b0, ka, boff0, boff1);
        q1 = ldfrag(b1, ka, boff0, boff1);
        q2 = ldfrag(b2, ka, boff0, boff1);
        q3 = ldfrag(b3, ka, boff0, boff1);
        WMMA4(a0, p0, p1, p2, p3)
        // prefetch tile kt+2 -> buffer 0, compute on buffer 1
        a0 = ldfrag(arow, kb, aoff0, aoff1);
        p0 = ldfrag(b0, kb, boff0, boff1);
        p1 = ldfrag(b1, kb, boff0, boff1);
        p2 = ldfrag(b2, kb, boff0, boff1);
        p3 = ldfrag(b3, kb, boff0, boff1);
        WMMA4(a1, q0, q1, q2, q3)
    }
    // peeled tail: tiles NK-2 (buffer 0) and NK-1 (buffer 1)
    {
        const int ka = (NK - 1) * 32;
        a1 = ldfrag(arow, ka, aoff0, aoff1);
        q0 = ldfrag(b0, ka, boff0, boff1);
        q1 = ldfrag(b1, ka, boff0, boff1);
        q2 = ldfrag(b2, ka, boff0, boff1);
        q3 = ldfrag(b3, ka, boff0, boff1);
        WMMA4(a0, p0, p1, p2, p3)
        WMMA4(a1, q0, q1, q2, q3)
    }
#undef WMMA4

    const int j = jt * 16 + mn;                  // 0..511 within hidden dim
    const float bi = biasg[j];
    const float bf = biasg[j + HDIM];
    const float bg = biasg[j + 2 * HDIM];
    const float bo = biasg[j + 3 * HDIM];

#pragma unroll
    for (int r = 0; r < 8; ++r) {
        const int row = mTile * 16 + r + 8 * khalf;   // C/D layout: m = r + 8*(lane/16)
        float iv = acc0[r] + bi;
        float fv = acc1[r] + bf;
        float gv = acc2[r] + bg;
        float ov = acc3[r] + bo;
        float cold = (step == 0) ? 0.f : Cst[row * HDIM + j];
        float cn = sigmf(fv) * cold + sigmf(iv) * tanhf(gv);
        Cst[row * HDIM + j] = cn;
        float hc = sigmf(ov) * tanhf(cn);
        if (j < DM) {
            float h = QN[row * DM + j] + hc;
            Hout[row * DM + j] = h;
            Xn[row * KTOT + DM + j] = (_Float16)h;   // h region of next-step operand
        }
    }
}

// ---------------------------------------------------------------------------
// Attention read-out: attn = softmax(h @ SG^T), r = attn @ SG -> Xnext r region
// ---------------------------------------------------------------------------
__global__ void __launch_bounds__(256) attn_kernel(
    const float* __restrict__ H, const float* __restrict__ SG,
    _Float16* __restrict__ Xn)
{
    __shared__ float sg[FEW][DM];
    const int t = threadIdx.x;
    for (int i = t; i < FEW * DM; i += 256) sg[i / DM][i % DM] = SG[i];
    __syncthreads();
    const int wid = t >> 5, lane = t & 31;
    const int row = blockIdx.x * 8 + wid;
    const float* hr = H + row * DM;
    const int cb = lane * 8;
    float d0 = 0.f, d1 = 0.f, d2 = 0.f, d3 = 0.f, d4 = 0.f;
#pragma unroll
    for (int i = 0; i < 8; ++i) {
        const int c = cb + i;
        float h = hr[c];
        d0 += h * sg[0][c]; d1 += h * sg[1][c]; d2 += h * sg[2][c];
        d3 += h * sg[3][c]; d4 += h * sg[4][c];
    }
#pragma unroll
    for (int off = 16; off; off >>= 1) {
        d0 += __shfl_xor(d0, off); d1 += __shfl_xor(d1, off);
        d2 += __shfl_xor(d2, off); d3 += __shfl_xor(d3, off);
        d4 += __shfl_xor(d4, off);
    }
    float m = fmaxf(fmaxf(fmaxf(d0, d1), fmaxf(d2, d3)), d4);
    float e0 = __expf(d0 - m), e1 = __expf(d1 - m), e2 = __expf(d2 - m);
    float e3 = __expf(d3 - m), e4 = __expf(d4 - m);
    float s = e0 + e1 + e2 + e3 + e4;
    float a0 = e0 / s, a1 = e1 / s, a2 = e2 / s, a3 = e3 / s, a4 = e4 / s;
    _Float16* xo = Xn + row * KTOT + 2 * DM + cb;
#pragma unroll
    for (int i = 0; i < 8; ++i) {
        const int c = cb + i;
        float r = a0 * sg[0][c] + a1 * sg[1][c] + a2 * sg[2][c]
                + a3 * sg[3][c] + a4 * sg[4][c];
        xo[i] = (_Float16)r;
    }
}

// ---------------------------------------------------------------------------
// Final: out[b] = h[b] . mean(support_g)
// ---------------------------------------------------------------------------
__global__ void __launch_bounds__(256) out_kernel(
    const float* __restrict__ H, const float* __restrict__ SM,
    float* __restrict__ out)
{
    const int wid = threadIdx.x >> 5, lane = threadIdx.x & 31;
    const int row = blockIdx.x * 8 + wid;
    const float* hr = H + row * DM;
    float acc = 0.f;
#pragma unroll
    for (int i = 0; i < 8; ++i) {
        const int c = lane * 8 + i;
        acc += hr[c] * SM[c];
    }
#pragma unroll
    for (int off = 16; off; off >>= 1) acc += __shfl_xor(acc, off);
    if (lane == 0) out[row] = acc;
}

// ---------------------------------------------------------------------------
extern "C" void kernel_launch(void* const* d_in, const int* in_sizes, int n_in,
                              void* d_out, int out_size, void* d_ws, size_t ws_size,
                              hipStream_t stream)
{
    (void)in_sizes; (void)n_in; (void)out_size; (void)ws_size;
    const int* q_l1 = (const int*)d_in[0];
    const int* q_l2 = (const int*)d_in[1];
    const int* q_r1 = (const int*)d_in[2];
    const int* q_r2 = (const int*)d_in[3];
    const int* s_l1 = (const int*)d_in[4];
    const int* s_l2 = (const int*)d_in[5];
    const int* s_r1 = (const int*)d_in[6];
    const int* s_r2 = (const int*)d_in[7];
    // d_in[8..11]: degree inputs, unused by the reference computation
    const float* emb      = (const float*)d_in[12];
    const float* gcn_w    = (const float*)d_in[13];
    const float* gcn_wb   = (const float*)d_in[14];
    const float* gcn_b    = (const float*)d_in[15];
    const float* hop_gate = (const float*)d_in[16];
    const float* attn_w   = (const float*)d_in[17];
    // d_in[18]: attn_b is softmax-invariant, unused
    const float* p1_w = (const float*)d_in[19];
    const float* p1_b = (const float*)d_in[20];
    const float* p2_w = (const float*)d_in[21];
    const float* p2_b = (const float*)d_in[22];
    const float* ln_g = (const float*)d_in[23];
    const float* ln_b = (const float*)d_in[24];
    const float* w_ih = (const float*)d_in[25];
    const float* w_hh = (const float*)d_in[26];
    const float* b_ih = (const float*)d_in[27];
    const float* b_hh = (const float*)d_in[28];

    char* wsp = (char*)d_ws;
    size_t off = 0;
    auto alloc = [&](size_t bytes) -> void* {
        void* p = wsp + off;
        off = (off + bytes + 255) & ~(size_t)255;
        return p;
    };
    float*    gwv   = (float*)alloc(2 * sizeof(float));
    float*    vvec  = (float*)alloc(DM * sizeof(float));
    float*    gbias = (float*)alloc(DE * sizeof(float));
    float*    biasg = (float*)alloc(NGATE * sizeof(float));
    _Float16* Wc    = (_Float16*)alloc((size_t)NGATE * KTOT * sizeof(_Float16));
    _Float16* Xa    = (_Float16*)alloc((size_t)B_ROWS * KTOT * sizeof(_Float16));
    _Float16* Xb    = (_Float16*)alloc((size_t)B_ROWS * KTOT * sizeof(_Float16));
    float*    QN    = (float*)alloc((size_t)B_ROWS * DM * sizeof(float));
    float*    SN    = (float*)alloc((size_t)FEW * DM * sizeof(float));
    float*    SG    = (float*)alloc((size_t)FEW * DM * sizeof(float));
    float*    SM    = (float*)alloc((size_t)DM * sizeof(float));
    float*    Cst   = (float*)alloc((size_t)B_ROWS * HDIM * sizeof(float));
    float*    Hbuf  = (float*)alloc((size_t)B_ROWS * DM * sizeof(float));

    prep_kernel<<<1, 256, 0, stream>>>(hop_gate, attn_w, gcn_w, gcn_wb, gcn_b,
                                       b_ih, b_hh, gwv, vvec, gbias, biasg);
    convw_kernel<<<(NGATE * KTOT + 255) / 256, 256, 0, stream>>>(w_ih, w_hh, Wc);
    neighbor_kernel<<<2 * (B_ROWS + FEW), 256, 0, stream>>>(
        q_l1, q_l2, q_r1, q_r2, s_l1, s_l2, s_r1, s_r2,
        emb, gcn_w, gbias, vvec, gwv, QN, SN, Xa, Xb);
    support_enc_kernel<<<1, 256, 0, stream>>>(SN, p1_w, p1_b, p2_w, p2_b,
                                              ln_g, ln_b, SG, SM);
    _Float16* Xc = Xa;
    _Float16* Xn = Xb;
    for (int step = 0; step < 4; ++step) {
        gemm_lstm_kernel<<<1024, 256, 0, stream>>>(Xc, Wc, biasg, Cst, QN,
                                                   Hbuf, Xn, step);
        if (step < 3)
            attn_kernel<<<512, 256, 0, stream>>>(Hbuf, SG, Xn);
        _Float16* tmp = Xc; Xc = Xn; Xn = tmp;
    }
    out_kernel<<<512, 256, 0, stream>>>(Hbuf, SM, (float*)d_out);
}
